// LocalRefinementUnit_54065048322153
// MI455X (gfx1250) — matrix operand order repaired and checked by reading, not compiled, not run on hardware
//
#include <hip/hip_runtime.h>
#include <math.h>

// ---------------- problem constants (match reference) ----------------
#define B_   4
#define C_   128
#define RN_  4096
#define K_   16
#define NROWS (B_*RN_*K_)   // 262144 rows for the MLPs
#define EPSV 1e-5f

// ---------------- types for WMMA ----------------
typedef __attribute__((ext_vector_type(16))) __bf16 v16bf;
typedef __attribute__((ext_vector_type(8)))  float  v8f;

struct __align__(16) U4 { unsigned int x, y, z, w; };
struct U8 { U4 lo, hi; };   // 32B = one v16bf fragment

__device__ __forceinline__ unsigned short f2bf(float f) {
    unsigned int u = __float_as_uint(f);
    u += 0x7FFFu + ((u >> 16) & 1u);            // round-to-nearest-even
    return (unsigned short)(u >> 16);
}

// =====================================================================
// 0) zero the global reduction buffers (harness poisons ws with 0xAA)
// =====================================================================
__global__ void k_init(float* st1, float* st2) {
    int t = threadIdx.x;
    if (t < 16)  st1[t] = 0.f;     // sub_Q first/second moments
    if (t < 256) st2[t] = 0.f;     // MLP2 per-channel sum / sumsq
}

// =====================================================================
// 1) KNN: points in LDS (48KB), register-resident sorted top-16
//    matches top_k(-d2) order: ascending (d2, index)
// =====================================================================
__global__ __launch_bounds__(128) void k_knn(const float* __restrict__ Qp,
                                             int* __restrict__ idx) {
    __shared__ float sx[RN_], sy[RN_], sz[RN_];
    int b = blockIdx.y;
    const float* Q = Qp + (size_t)b * 3 * RN_;
    for (int i = threadIdx.x; i < RN_; i += 128) {
        sx[i] = Q[i]; sy[i] = Q[RN_ + i]; sz[i] = Q[2 * RN_ + i];
    }
    __syncthreads();

    int q = blockIdx.x * 128 + threadIdx.x;
    float qx = sx[q], qy = sy[q], qz = sz[q];

    float bd[K_]; int bi[K_];
#pragma unroll
    for (int j = 0; j < K_; ++j) { bd[j] = 3.4e38f; bi[j] = 0x7fffffff; }

    for (int j = 0; j < RN_; ++j) {
        float dx = qx - sx[j], dy = qy - sy[j], dz = qz - sz[j];
        float d = fmaf(dz, dz, fmaf(dy, dy, dx * dx));
        if (d < bd[K_ - 1] || (d == bd[K_ - 1] && j < bi[K_ - 1])) {
#pragma unroll
            for (int t = K_ - 1; t >= 0; --t) {
                bool ltp = (t == 0) ? false
                         : ((d < bd[t - 1]) || (d == bd[t - 1] && j < bi[t - 1]));
                bool ltc = (d < bd[t]) || (d == bd[t] && j < bi[t]);
                float nd = ltp ? bd[t - 1] : (ltc ? d : bd[t]);
                int   ni = ltp ? bi[t - 1] : (ltc ? j : bi[t]);
                bd[t] = nd; bi[t] = ni;
            }
        }
    }
    int* op = idx + ((size_t)(b * RN_ + q)) * K_;
#pragma unroll
    for (int j = 0; j < K_; ++j) op[j] = bi[j];
}

// =====================================================================
// 2) F_E (B,C,RN) fp32 -> Fbf (B,RN,C) bf16  (row-gatherable layout)
// =====================================================================
__global__ void k_fbf(const float* __restrict__ FE, unsigned short* __restrict__ Fbf) {
    size_t i = (size_t)blockIdx.x * 256 + threadIdx.x;   // (b*C+c)*RN + r
    int r = (int)(i & (RN_ - 1));
    size_t bc = i >> 12;
    int c = (int)(bc & (C_ - 1));
    int b = (int)(bc >> 7);
    Fbf[((size_t)(b * RN_ + r)) * C_ + c] = f2bf(FE[i]);
}

// W2 (128x192) fp32 -> bf16
__global__ void k_w2bf(const float* __restrict__ W2, unsigned short* __restrict__ W2bf) {
    int i = blockIdx.x * 256 + threadIdx.x;
    if (i < 128 * 192) W2bf[i] = f2bf(W2[i]);
}

// =====================================================================
// 3) first/second moments of sub_Q over all 262144 rows (9 floats)
// =====================================================================
__global__ __launch_bounds__(256) void k_stats1(const float* __restrict__ Qp,
                                                const int* __restrict__ idx,
                                                float* __restrict__ st) {
    __shared__ float acc[9];
    if (threadIdx.x < 9) acc[threadIdx.x] = 0.f;
    __syncthreads();
    int gr = blockIdx.x * 256 + threadIdx.x;
    int G = gr >> 4, b = G >> 12, r = G & (RN_ - 1);
    int j = idx[gr];
    const float* Q = Qp + (size_t)b * 3 * RN_;
    float sx = Q[j] - Q[r], sy = Q[RN_ + j] - Q[RN_ + r], sz = Q[2 * RN_ + j] - Q[2 * RN_ + r];
    float v[9] = { sx, sy, sz, sx * sx, sx * sy, sx * sz, sy * sy, sy * sz, sz * sz };
#pragma unroll
    for (int t = 0; t < 9; ++t) {
        float x = v[t];
#pragma unroll
        for (int o = 16; o > 0; o >>= 1) x += __shfl_xor(x, o, 32);
        if ((threadIdx.x & 31) == 0) atomicAdd(&acc[t], x);
    }
    __syncthreads();
    if (threadIdx.x < 9) atomicAdd(&st[threadIdx.x], acc[threadIdx.x]);
}

// =====================================================================
// 4) fold BN analytically into MLP1 / MLPw effective weights
//    mu_c = W_c.m + b_c ; var_c = W_c Cov W_c^T ; W'=aW ; b'=be - a*(W.m)
// =====================================================================
__global__ void k_params1(const float* __restrict__ st,
                          const float* __restrict__ W1, const float* __restrict__ g1, const float* __restrict__ be1,
                          const float* __restrict__ Ww, const float* __restrict__ gw, const float* __restrict__ bew,
                          float* __restrict__ P1, float* __restrict__ PW) {
    int t = threadIdx.x;
    if (t >= 80) return;
    float invN = 1.0f / (float)NROWS;
    float m0 = st[0] * invN, m1 = st[1] * invN, m2 = st[2] * invN;
    float Cm[3][3];
    Cm[0][0] = st[3] * invN - m0 * m0; Cm[0][1] = st[4] * invN - m0 * m1; Cm[0][2] = st[5] * invN - m0 * m2;
    Cm[1][1] = st[6] * invN - m1 * m1; Cm[1][2] = st[7] * invN - m1 * m2; Cm[2][2] = st[8] * invN - m2 * m2;
    Cm[1][0] = Cm[0][1]; Cm[2][0] = Cm[0][2]; Cm[2][1] = Cm[1][2];

    const float *W, *g, *be; float *Wo, *bo; int c;
    if (t < 64) { c = t;      W = W1 + 3 * c; g = g1; be = be1; Wo = P1 + 3 * c; bo = P1 + 192; }
    else        { c = t - 64; W = Ww + 3 * c; g = gw; be = bew; Wo = PW + 3 * c; bo = PW + 48;  }

    float w[3] = { W[0], W[1], W[2] };
    float var = 0.f;
#pragma unroll
    for (int i = 0; i < 3; ++i)
#pragma unroll
        for (int j = 0; j < 3; ++j) var += w[i] * w[j] * Cm[i][j];
    float a = g[c] / sqrtf(var + EPSV);
    Wo[0] = a * w[0]; Wo[1] = a * w[1]; Wo[2] = a * w[2];
    bo[c] = be[c] - a * (w[0] * m0 + w[1] * m1 + w[2] * m2);
}

// =====================================================================
// 5) encode: enc_sub = relu(s @ W1'^T + b1') -> bf16, and w_diag[k]
// =====================================================================
__global__ __launch_bounds__(256) void k_encode(const float* __restrict__ Qp,
                                                const int* __restrict__ idx,
                                                const float* __restrict__ P1,
                                                const float* __restrict__ PW,
                                                unsigned short* __restrict__ encA,
                                                float* __restrict__ wdiag) {
    int gr = blockIdx.x * 256 + threadIdx.x;
    int G = gr >> 4, k = gr & 15, b = G >> 12, r = G & (RN_ - 1);
    int j = idx[gr];
    const float* Q = Qp + (size_t)b * 3 * RN_;
    float sx = Q[j] - Q[r], sy = Q[RN_ + j] - Q[RN_ + r], sz = Q[2 * RN_ + j] - Q[2 * RN_ + r];

    float wv = fmaf(PW[3 * k + 2], sz, fmaf(PW[3 * k + 1], sy, fmaf(PW[3 * k], sx, PW[48 + k])));
    wdiag[gr] = fmaxf(wv, 0.f);

    unsigned int* eo = (unsigned int*)encA + (size_t)gr * 32;
#pragma unroll
    for (int c = 0; c < 64; c += 2) {
        float e0 = fmaxf(fmaf(P1[3 * c + 2], sz, fmaf(P1[3 * c + 1], sy, fmaf(P1[3 * c], sx, P1[192 + c]))), 0.f);
        float e1 = fmaxf(fmaf(P1[3 * c + 5], sz, fmaf(P1[3 * c + 4], sy, fmaf(P1[3 * c + 3], sx, P1[192 + c + 1]))), 0.f);
        eo[c >> 1] = (unsigned)f2bf(e0) | ((unsigned)f2bf(e1) << 16);
    }
}

// =====================================================================
// 6) MLP2 GEMM via v_wmma_f32_16x16x32_bf16.
//    One wave = one (b,r): its 16 neighbors = the WMMA M=16 tile.
//    K=192 (6 chunks of 32), N=128 (8 tiles of 16). Weights in LDS
//    (padded stride 200 -> conflict-free column reads). A gathered
//    straight from L2-resident bf16 buffers with global_load_b128.
//    PASS 1: per-channel sum/sumsq for BN.  PASS 2: BN+ReLU+w_diag
//    weighted k-reduction fused, + residual add, direct (B,C,RN) store.
// =====================================================================
template <int PASS>
__global__ __launch_bounds__(256) void k_gemm(const unsigned short* __restrict__ W2bf,
                                              const float* __restrict__ b2,
                                              const float* __restrict__ bn2,
                                              const unsigned short* __restrict__ Fbf,
                                              const unsigned short* __restrict__ encA,
                                              const int* __restrict__ idx,
                                              const float* __restrict__ wdiag,
                                              const float* __restrict__ FE,
                                              float* __restrict__ out,
                                              float* __restrict__ stats2) {
    constexpr int WS = 200;                       // padded bf16 stride
    __shared__ __align__(16) unsigned short sW[128 * WS];
    __shared__ float sSum[128], sSq[128];
    int tid = threadIdx.x;
    if (PASS == 1 && tid < 128) { sSum[tid] = 0.f; sSq[tid] = 0.f; }
    {
        const unsigned int* src = (const unsigned int*)W2bf;
        unsigned int* dst = (unsigned int*)sW;
        for (int i = tid; i < 128 * 96; i += 256) {
            int row = i / 96, col = i - row * 96;
            dst[row * (WS / 2) + col] = src[i];
        }
    }
    __syncthreads();

    int wave = tid >> 5, lane = tid & 31, half = lane >> 4, lm = lane & 15;
    int G = blockIdx.x * 8 + wave;                // (b,r) group
    int b = G >> 12, r = G & (RN_ - 1);
    int gbase = G << 4;                           // first of 16 rows

    int jn = idx[gbase + lm];
    const U4* fp = (const U4*)(Fbf + ((size_t)b * RN_ + jn) * C_);   // 16 U4 = 128 bf16
    const U4* ep = (const U4*)(encA + (size_t)(gbase + lm) * 64);    // 8  U4 = 64 bf16

    // A fragments: lane half 0 holds k in {0..7,16..23} of each 32-chunk,
    // half 1 holds {8..15,24..31} (per 16-bit A-matrix layout).
    v16bf A[6];
#pragma unroll
    for (int kc = 0; kc < 6; ++kc) {
        U8 f;
        if (kc < 4) { int u = kc * 4 + half; f.lo = fp[u]; f.hi = fp[u + 2]; }
        else        { int u = (kc - 4) * 4 + half; f.lo = ep[u]; f.hi = ep[u + 2]; }
        A[kc] = __builtin_bit_cast(v16bf, f);
    }

    const U4* sW4 = (const U4*)sW;
    v8f acc[8];
#pragma unroll
    for (int nt = 0; nt < 8; ++nt) acc[nt] = (v8f){0.f, 0.f, 0.f, 0.f, 0.f, 0.f, 0.f, 0.f};

#pragma unroll
    for (int nt = 0; nt < 8; ++nt) {
#pragma unroll
        for (int kc = 0; kc < 6; ++kc) {
            // B: lane = column n, half 0 holds k {0..7,16..23}, half 1 {8..15,24..31}
            int o = (nt * 16 + lm) * (WS / 8) + kc * 4 + half;
            U8 f; f.lo = sW4[o]; f.hi = sW4[o + 2];
            v16bf Bf = __builtin_bit_cast(v16bf, f);
            acc[nt] = __builtin_amdgcn_wmma_f32_16x16x32_bf16(
                false, A[kc], false, Bf, (short)0, acc[nt], false, false);
        }
    }

    if (PASS == 1) {
#pragma unroll
        for (int nt = 0; nt < 8; ++nt) {
            int c = nt * 16 + lm;
            float bc = b2[c], s1 = 0.f, s2 = 0.f;
#pragma unroll
            for (int j = 0; j < 8; ++j) { float y = acc[nt][j] + bc; s1 += y; s2 = fmaf(y, y, s2); }
            s1 += __shfl_xor(s1, 16, 32);
            s2 += __shfl_xor(s2, 16, 32);
            if (!half) { atomicAdd(&sSum[c], s1); atomicAdd(&sSq[c], s2); }
        }
        __syncthreads();
        if (tid < 128) { atomicAdd(&stats2[tid], sSum[tid]); atomicAdd(&stats2[128 + tid], sSq[tid]); }
    } else {
        float wme = wdiag[gbase + lm];            // lane lm holds w_diag of row lm
        float wdl[8];
#pragma unroll
        for (int j = 0; j < 8; ++j) wdl[j] = __shfl(wme, j + 8 * half, 32);
#pragma unroll
        for (int nt = 0; nt < 8; ++nt) {
            int c = nt * 16 + lm;
            float a2 = bn2[c], bb = bn2[128 + c], p = 0.f;
#pragma unroll
            for (int j = 0; j < 8; ++j) {
                float h = fmaxf(fmaf(a2, acc[nt][j], bb), 0.f);
                p = fmaf(h, wdl[j], p);
            }
            p += __shfl_xor(p, 16, 32);           // combine M 0..7 with 8..15
            if (!half) {
                size_t o = ((size_t)(b * C_) + c) * RN_ + r;
                out[o] = p + FE[o];               // residual, already (B,C,RN)
            }
        }
    }
}

// =====================================================================
// 7) MLP2 BN scale/shift from the accumulated stats
// =====================================================================
__global__ void k_bn2(const float* __restrict__ st2, const float* __restrict__ b2,
                      const float* __restrict__ g2, const float* __restrict__ be2,
                      float* __restrict__ bn2) {
    int c = threadIdx.x;
    if (c >= 128) return;
    float invN = 1.0f / (float)NROWS;
    float mu = st2[c] * invN;
    float var = st2[128 + c] * invN - mu * mu;
    float a = g2[c] / sqrtf(var + EPSV);
    bn2[c] = a;
    bn2[128 + c] = fmaf(a, b2[c] - mu, be2[c]);
}

// =====================================================================
// launcher
// =====================================================================
extern "C" void kernel_launch(void* const* d_in, const int* in_sizes, int n_in,
                              void* d_out, int out_size, void* d_ws, size_t ws_size,
                              hipStream_t stream) {
    (void)in_sizes; (void)n_in; (void)out_size; (void)ws_size;
    const float* F_E = (const float*)d_in[0];
    const float* Qp  = (const float*)d_in[1];
    const float* W1  = (const float*)d_in[2];
    // b1 (d_in[3]) cancels analytically in the folded bias
    const float* g1  = (const float*)d_in[4];
    const float* be1 = (const float*)d_in[5];
    const float* W2  = (const float*)d_in[6];
    const float* b2  = (const float*)d_in[7];
    const float* g2  = (const float*)d_in[8];
    const float* be2 = (const float*)d_in[9];
    const float* Ww  = (const float*)d_in[10];
    const float* gw  = (const float*)d_in[12];
    const float* bew = (const float*)d_in[13];
    float* out = (float*)d_out;

    char* ws = (char*)d_ws;
    int*            p_idx  = (int*)(ws);                              // 1 MB
    float*          p_wd   = (float*)(ws + (1u << 20));               // 1 MB
    float*          p_st1  = (float*)(ws + (2u << 20));               // 16 f
    float*          p_st2  = (float*)(ws + (2u << 20) + 256);         // 256 f
    float*          p_bn2  = (float*)(ws + (2u << 20) + 2048);        // 256 f
    float*          p_P1   = (float*)(ws + (2u << 20) + 4096);        // 256 f
    float*          p_PW   = (float*)(ws + (2u << 20) + 8192);        // 64 f
    unsigned short* p_w2bf = (unsigned short*)(ws + (2u << 20) + 16384);  // 48 KB
    unsigned short* p_fbf  = (unsigned short*)(ws + (3u << 20));      // 4 MB
    unsigned short* p_enc  = (unsigned short*)(ws + (7u << 20));      // 32 MB

    k_init<<<1, 512, 0, stream>>>(p_st1, p_st2);
    k_knn<<<dim3(RN_ / 128, B_), 128, 0, stream>>>(Qp, p_idx);
    k_w2bf<<<96, 256, 0, stream>>>(W2, p_w2bf);
    k_fbf<<<(B_ * C_ * RN_) / 256, 256, 0, stream>>>(F_E, p_fbf);
    k_stats1<<<NROWS / 256, 256, 0, stream>>>(Qp, p_idx, p_st1);
    k_params1<<<1, 128, 0, stream>>>(p_st1, W1, g1, be1, Ww, gw, bew, p_P1, p_PW);
    k_encode<<<NROWS / 256, 256, 0, stream>>>(Qp, p_idx, p_P1, p_PW, p_enc, p_wd);
    k_gemm<1><<<NROWS / 128, 256, 0, stream>>>(p_w2bf, b2, p_bn2, p_fbf, p_enc,
                                               p_idx, p_wd, F_E, out, p_st2);
    k_bn2<<<1, 128, 0, stream>>>(p_st2, b2, g2, be2, p_bn2);
    k_gemm<2><<<NROWS / 128, 256, 0, stream>>>(p_w2bf, b2, p_bn2, p_fbf, p_enc,
                                               p_idx, p_wd, F_E, out, p_st2);
}
